// single_semantic_50208167690950
// MI455X (gfx1250) — compile-verified, matrix operand order checked by reading, not run on hardware
//
#include <hip/hip_runtime.h>
#include <hip/hip_bf16.h>

#define E_ 65536
#define L_ 3
#define N_ 20000
#define T_ 16384
#define D_ 128
#define H_ 8
#define AH_ 4
#define HD_ 32
#define FF_ 2048

#define QS_ 56  // column stride (elements) of column-major qkv LDS buffer

typedef __attribute__((ext_vector_type(8)))  __bf16 v8bf;
typedef __attribute__((ext_vector_type(16))) __bf16 v16bf;
typedef __attribute__((ext_vector_type(8)))  float  v8f;

// ---- WMMA helpers (CDNA5 v_wmma_f32_16x16x32_bf16, wave32) -----------------

// A (16x32 MxK) / B (32x16 KxN as W rows = B columns) fragment load from a
// row-major bf16 matrix with leading dim `ld`.
// lane<16 : row lane,    K = k0+[0..8) and k0+[16..24)
// lane>=16: row lane-16, K = k0+8+[0..8) and k0+24+[0..8)
__device__ __forceinline__ v16bf frag_ld(const __bf16* p, int ld, int lane, int k0) {
  int r  = lane & 15;
  int kb = k0 + ((lane >> 4) << 3);
  const __bf16* q = p + r * ld + kb;
  v8bf c0 = *(const v8bf*)(q);
  v8bf c1 = *(const v8bf*)(q + 16);
  v16bf a;
#pragma unroll
  for (int i = 0; i < 8; ++i) { a[i] = c0[i]; a[i + 8] = c1[i]; }
  return a;
}

__device__ __forceinline__ v8f wmma_bf16(v16bf a, v16bf b, v8f c) {
  return __builtin_amdgcn_wmma_f32_16x16x32_bf16(false, a, false, b, (short)0, c,
                                                 false, false);
}

__device__ __forceinline__ float wave_sum(float v) {
#pragma unroll
  for (int m = 16; m >= 1; m >>= 1) v += __shfl_xor(v, m, 32);
  return v;
}

// order-preserving float<->uint encoding for atomicMax-based segment max
__device__ __forceinline__ unsigned fenc(float f) {
  unsigned u = __float_as_uint(f);
  return (u & 0x80000000u) ? ~u : (u | 0x80000000u);
}
__device__ __forceinline__ float fdec(unsigned u) {
  return (u & 0x80000000u) ? __uint_as_float(u ^ 0x80000000u) : __uint_as_float(~u);
}

// ---- K0: fp32 -> bf16 conversion -------------------------------------------
__global__ __launch_bounds__(256) void f2bf_kernel(const float* __restrict__ s,
                                                   __bf16* __restrict__ d, int n) {
  int i = blockIdx.x * 256 + threadIdx.x;
  if (i < n) d[i] = (__bf16)s[i];
}

// ---- K1: per-edge transformer encoder -> hidden (E x 128) ------------------
// 16 edges / block, M = 48 rows, 8 waves.
__global__ __launch_bounds__(256) void encoder_kernel(
    const __bf16* __restrict__ featB, const int* __restrict__ emi,
    const __bf16* __restrict__ wqkvB, const float* __restrict__ in_b,
    const __bf16* __restrict__ woB,  const float* __restrict__ out_b,
    const float* __restrict__ ln1g, const float* __restrict__ ln1b,
    const __bf16* __restrict__ w1B, const float* __restrict__ b1,
    const __bf16* __restrict__ w2B, const float* __restrict__ b2,
    const float* __restrict__ ln2g, const float* __restrict__ ln2b,
    float* __restrict__ hidden, __bf16* __restrict__ hiddenB) {
  __shared__ __bf16 sX[48 * 128];     // token activations (row-major, A operand)
  __shared__ __bf16 sQKV[384 * QS_];  // qkv, COLUMN-major: [col*QS_ + row]
  __shared__ float  sF[48 * 128];     // fp32 scratch (proj / ffn out)
  __shared__ float  sX1[48 * 128];    // fp32 x after LN1 (residual 2)
  __shared__ __bf16 sH[48 * 128];     // attn-out / ffn-hidden chunk (row-major)
  __shared__ int    sIdx[48];

  const int tid = threadIdx.x;
  const int wave = tid >> 5, lane = tid & 31;
  const int e0 = blockIdx.x * 16;

  if (tid < 48) sIdx[tid] = emi[(e0 + tid / 3) * 3 + (tid % 3)];
  __syncthreads();

  // gather x = bf16(features[idx]) : 48 rows x 128
  for (int c = tid; c < 48 * 16; c += 256) {
    int r = c >> 4, o = (c & 15) << 3;
    *(v8bf*)(&sX[r * 128 + o]) = *(const v8bf*)(featB + (size_t)sIdx[r] * 128 + o);
  }
  __syncthreads();

  // in_proj: qkv = x @ Wqkv^T + b   (M=48, N=384, K=128)
  // 9 tiles/wave, statically unrolled in groups of 3 so independent tiles
  // interleave (hides WMMA latency / hazard NOPs).
  for (int ii = 0; ii < 9; ii += 3) {
#pragma unroll
    for (int i = 0; i < 3; ++i) {
      int t = wave + (ii + i) * 8;
      int mt = t / 24, nt = t % 24;
      v8f c = {};
#pragma unroll
      for (int k0 = 0; k0 < 128; k0 += 32) {
        v16bf a = frag_ld(&sX[mt * 16 * 128], 128, lane, k0);
        v16bf b = frag_ld(wqkvB + nt * 16 * 128, 128, lane, k0);
        c = wmma_bf16(a, b, c);
      }
      int n  = nt * 16 + (lane & 15);
      int mb = mt * 16 + ((lane >> 4) << 3);
      float bias = in_b[n];
      v8bf pk;
#pragma unroll
      for (int j = 0; j < 8; ++j) pk[j] = (__bf16)(c[j] + bias);
      *(v8bf*)(&sQKV[n * QS_ + mb]) = pk;
    }
  }
  __syncthreads();

  // tiny attention: per (edge, head, l): 192 threads; L=3, AH=4, HD=32
  if (tid < 192) {
    int p = tid / 3, l = tid % 3;
    int le = p >> 2, h = p & 3;
    int tr = le * 3;                 // base token row of this edge
    int qc = h * 32;                 // q column base; k: +128; v: +256
    float sc[3];
#pragma unroll
    for (int m = 0; m < 3; ++m) {
      float d = 0.f;
      for (int t = 0; t < 32; ++t)
        d += (float)sQKV[(qc + t) * QS_ + tr + l] *
             (float)sQKV[(128 + qc + t) * QS_ + tr + m];
      sc[m] = d * 0.1767766952966369f;  // 1/sqrt(32)
    }
    float mx = fmaxf(sc[0], fmaxf(sc[1], sc[2]));
    float a0 = expf(sc[0] - mx), a1 = expf(sc[1] - mx), a2 = expf(sc[2] - mx);
    float inv = 1.f / (a0 + a1 + a2);
    a0 *= inv; a1 *= inv; a2 *= inv;
    for (int t = 0; t < 32; ++t) {
      float o = a0 * (float)sQKV[(256 + qc + t) * QS_ + tr + 0] +
                a1 * (float)sQKV[(256 + qc + t) * QS_ + tr + 1] +
                a2 * (float)sQKV[(256 + qc + t) * QS_ + tr + 2];
      sH[(tr + l) * 128 + qc + t] = (__bf16)o;
    }
  }
  __syncthreads();

  // out_proj: sF = o @ Wo^T + b   (M=48, N=128, K=128), 3 tiles/wave unrolled
#pragma unroll
  for (int i = 0; i < 3; ++i) {
    int t = wave + i * 8;
    int mt = t / 8, nt = t % 8;
    v8f c = {};
#pragma unroll
    for (int k0 = 0; k0 < 128; k0 += 32) {
      v16bf a = frag_ld(&sH[mt * 16 * 128], 128, lane, k0);
      v16bf b = frag_ld(woB + nt * 16 * 128, 128, lane, k0);
      c = wmma_bf16(a, b, c);
    }
    int n  = nt * 16 + (lane & 15);
    int mb = mt * 16 + ((lane >> 4) << 3);
    float bias = out_b[n];
#pragma unroll
    for (int j = 0; j < 8; ++j) sF[(mb + j) * 128 + n] = c[j] + bias;
  }
  __syncthreads();

  // LN1: x1 = LN(x + o_proj); keep fp32 in sX1, bf16 in sX
  for (int r = wave; r < 48; r += 8) {
    float v[4], s = 0.f;
#pragma unroll
    for (int i = 0; i < 4; ++i) {
      int cix = lane + 32 * i;
      float xv = (float)sX[r * 128 + cix] + sF[r * 128 + cix];
      v[i] = xv; s += xv;
    }
    s = wave_sum(s);
    float mu = s * (1.f / 128.f), q = 0.f;
#pragma unroll
    for (int i = 0; i < 4; ++i) { float d2 = v[i] - mu; q += d2 * d2; }
    q = wave_sum(q);
    float rs = rsqrtf(q * (1.f / 128.f) + 1e-5f);
#pragma unroll
    for (int i = 0; i < 4; ++i) {
      int cix = lane + 32 * i;
      float y = (v[i] - mu) * rs * ln1g[cix] + ln1b[cix];
      sX1[r * 128 + cix] = y;
      sX[r * 128 + cix]  = (__bf16)y;
    }
  }
  __syncthreads();

  // fused FFN: loop FF in chunks of 128; per-wave register accumulators
  v8f acc[3] = {};
  for (int f0 = 0; f0 < FF_; f0 += 128) {
    // prefetch next chunk's weights into cache (global_prefetch_b8)
    if (f0 + 128 < FF_) {
      __builtin_prefetch(w1B + (f0 + 128) * 128 + tid * 64, 0, 3);
      int rr = tid >> 1, oo = (tid & 1) << 6;
      __builtin_prefetch(w2B + rr * 2048 + (f0 + 128) + oo, 0, 3);
    }
    // GEMM1 + relu: h = relu(x1 @ W1[f0:f0+128]^T + b1), 3 tiles/wave unrolled
#pragma unroll
    for (int i = 0; i < 3; ++i) {
      int t = wave + i * 8;
      int mt = t / 8, nt = t % 8;
      v8f c = {};
#pragma unroll
      for (int k0 = 0; k0 < 128; k0 += 32) {
        v16bf a = frag_ld(&sX[mt * 16 * 128], 128, lane, k0);
        v16bf b = frag_ld(w1B + (f0 + nt * 16) * 128, 128, lane, k0);
        c = wmma_bf16(a, b, c);
      }
      int n  = nt * 16 + (lane & 15);
      int mb = mt * 16 + ((lane >> 4) << 3);
      float bias = b1[f0 + n];
#pragma unroll
      for (int j = 0; j < 8; ++j)
        sH[(mb + j) * 128 + n] = (__bf16)fmaxf(c[j] + bias, 0.f);
    }
    __syncthreads();
    // GEMM2: acc += h @ W2[:, f0:f0+128]^T   (B col n = w2 row n, ld=2048)
#pragma unroll
    for (int i = 0; i < 3; ++i) {
      int t = wave * 3 + i;
      int mt = t / 8, nt = t % 8;
#pragma unroll
      for (int k0 = 0; k0 < 128; k0 += 32) {
        v16bf a = frag_ld(&sH[mt * 16 * 128], 128, lane, k0);
        v16bf b = frag_ld(w2B + nt * 16 * 2048 + f0, 2048, lane, k0);
        acc[i] = wmma_bf16(a, b, acc[i]);
      }
    }
    __syncthreads();
  }
#pragma unroll
  for (int i = 0; i < 3; ++i) {
    int t = wave * 3 + i;
    int mt = t / 8, nt = t % 8;
    int n  = nt * 16 + (lane & 15);
    int mb = mt * 16 + ((lane >> 4) << 3);
    float bias = b2[n];
#pragma unroll
    for (int j = 0; j < 8; ++j) sF[(mb + j) * 128 + n] = acc[i][j] + bias;
  }
  __syncthreads();

  // LN2: x2 = LN(x1 + ffn) -> sF (fp32)
  for (int r = wave; r < 48; r += 8) {
    float v[4], s = 0.f;
#pragma unroll
    for (int i = 0; i < 4; ++i) {
      int cix = lane + 32 * i;
      float xv = sX1[r * 128 + cix] + sF[r * 128 + cix];
      v[i] = xv; s += xv;
    }
    s = wave_sum(s);
    float mu = s * (1.f / 128.f), q = 0.f;
#pragma unroll
    for (int i = 0; i < 4; ++i) { float d2 = v[i] - mu; q += d2 * d2; }
    q = wave_sum(q);
    float rs = rsqrtf(q * (1.f / 128.f) + 1e-5f);
#pragma unroll
    for (int i = 0; i < 4; ++i) {
      int cix = lane + 32 * i;
      sF[r * 128 + cix] = (v[i] - mu) * rs * ln2g[cix] + ln2b[cix];
    }
  }
  __syncthreads();

  // hidden = mean over L tokens
  for (int c = tid; c < 16 * 128; c += 256) {
    int le = c >> 7, d = c & 127;
    int e = e0 + le;
    float h3 = (sF[(le * 3 + 0) * 128 + d] + sF[(le * 3 + 1) * 128 + d] +
                sF[(le * 3 + 2) * 128 + d]) * (1.f / 3.f);
    hidden[(size_t)e * 128 + d]  = h3;
    hiddenB[(size_t)e * 128 + d] = (__bf16)h3;
  }
}

// ---- K2: eft = hidden @ rnn_w^T + rnn_b  (M=E, N=1024, K=128) --------------
__global__ __launch_bounds__(256) void rnn_kernel(
    const __bf16* __restrict__ hiddenB, const __bf16* __restrict__ wrB,
    const float* __restrict__ rnn_b, float* __restrict__ eft) {
  __shared__ __bf16 sA[16 * 128];  // A tile shared by all 8 waves
  int tid = threadIdx.x;
  int wave = tid >> 5, lane = tid & 31;
  int mt = blockIdx.x;  // E/16 blocks
  {
    int r = tid >> 4, o = (tid & 15) << 3;
    *(v8bf*)(&sA[r * 128 + o]) =
        *(const v8bf*)(hiddenB + (size_t)mt * 16 * 128 + r * 128 + o);
  }
  __syncthreads();
  v8f acc[8] = {};
  for (int k0 = 0; k0 < 128; k0 += 32) {
    v16bf a = frag_ld(sA, 128, lane, k0);  // reused across 8 N-tiles
#pragma unroll
    for (int i = 0; i < 8; ++i) {
      v16bf b = frag_ld(wrB + (wave * 8 + i) * 16 * 128, 128, lane, k0);
      acc[i] = wmma_bf16(a, b, acc[i]);
    }
  }
  int mb = mt * 16 + ((lane >> 4) << 3);
#pragma unroll
  for (int i = 0; i < 8; ++i) {
    int n = (wave * 8 + i) * 16 + (lane & 15);
    float bias = rnn_b[n];
#pragma unroll
    for (int j = 0; j < 8; ++j)
      eft[(size_t)(mb + j) * 1024 + n] = acc[i][j] + bias;
  }
}

// ---- K3: a[e,h] = center.attn1[h] + <eft[e,h], attn2[h]>, leaky, seg-max ---
__global__ __launch_bounds__(256) void attn_score_kernel(
    const float* __restrict__ features, const int* __restrict__ emi,
    const int* __restrict__ dst, const float* __restrict__ eft,
    const float* __restrict__ attn1, const float* __restrict__ attn2,
    float* __restrict__ aE, unsigned* __restrict__ mEnc) {
  int wave = threadIdx.x >> 5, lane = threadIdx.x & 31;
  int gw = blockIdx.x * 8 + wave;  // (e,h) pair
  int e = gw >> 3, h = gw & 7;
  int c = emi[e * 3 + 2];
  float4 fv = ((const float4*)(features + (size_t)c * 128))[lane];
  float4 wv = ((const float4*)(attn1 + h * 128))[lane];
  float4 ev = ((const float4*)(eft + (size_t)e * 1024 + h * 128))[lane];
  float4 a2 = ((const float4*)(attn2 + h * 128))[lane];
  float p = fv.x * wv.x + fv.y * wv.y + fv.z * wv.z + fv.w * wv.w +
            ev.x * a2.x + ev.y * a2.y + ev.z * a2.z + ev.w * a2.w;
  p = wave_sum(p);
  if (lane == 0) {
    float a = p > 0.f ? p : 0.01f * p;
    aE[gw] = a;
    atomicMax(&mEnc[dst[e] * 8 + h], fenc(a));
  }
}

// ---- K4: e = exp(a - m); segment sum ---------------------------------------
__global__ __launch_bounds__(256) void exp_sum_kernel(
    const float* __restrict__ aE, const int* __restrict__ dst,
    const unsigned* __restrict__ mEnc, float* __restrict__ eExp,
    float* __restrict__ sN) {
  int i = blockIdx.x * 256 + threadIdx.x;
  if (i >= E_ * H_) return;
  int e = i >> 3, h = i & 7;
  int n = dst[e];
  float ev = expf(aE[i] - fdec(mEnc[n * 8 + h]));
  eExp[i] = ev;
  atomicAdd(&sN[n * 8 + h], ev);
}

// ---- K5: ft[dst] += eft * (e / s) ------------------------------------------
__global__ __launch_bounds__(256) void scatter_kernel(
    const float* __restrict__ eExp, const float* __restrict__ sN,
    const int* __restrict__ dst, const float* __restrict__ eft,
    float* __restrict__ ft) {
  int wave = threadIdx.x >> 5, lane = threadIdx.x & 31;
  int gw = blockIdx.x * 8 + wave;
  int e = gw >> 3, h = gw & 7;
  int n = dst[e];
  float w = eExp[gw] / sN[n * 8 + h];
  float4 ev = ((const float4*)(eft + (size_t)e * 1024 + h * 128))[lane];
  float* p = ft + (size_t)n * 1024 + h * 128 + lane * 4;
  atomicAdd(p + 0, ev.x * w);
  atomicAdd(p + 1, ev.y * w);
  atomicAdd(p + 2, ev.z * w);
  atomicAdd(p + 3, ev.w * w);
}

// ---- K6: out = ft[target_idx] ----------------------------------------------
__global__ __launch_bounds__(256) void gather_out_kernel(
    const float* __restrict__ ft, const int* __restrict__ tgt,
    float* __restrict__ out) {
  int i = blockIdx.x * 256 + threadIdx.x;  // T*1024
  int t = i >> 10, j = i & 1023;
  out[i] = ft[(size_t)tgt[t] * 1024 + j];
}

// ---- host ------------------------------------------------------------------
extern "C" void kernel_launch(void* const* d_in, const int* in_sizes, int n_in,
                              void* d_out, int out_size, void* d_ws, size_t ws_size,
                              hipStream_t stream) {
  const float* features = (const float*)d_in[0];
  const int*   emi      = (const int*)d_in[1];
  const int*   dst      = (const int*)d_in[2];
  const int*   tgt      = (const int*)d_in[3];
  const float* in_w     = (const float*)d_in[4];
  const float* in_b     = (const float*)d_in[5];
  const float* out_w    = (const float*)d_in[6];
  const float* out_b    = (const float*)d_in[7];
  const float* ln1g     = (const float*)d_in[8];
  const float* ln1b     = (const float*)d_in[9];
  const float* w1       = (const float*)d_in[10];
  const float* b1       = (const float*)d_in[11];
  const float* w2       = (const float*)d_in[12];
  const float* b2       = (const float*)d_in[13];
  const float* ln2g     = (const float*)d_in[14];
  const float* ln2b     = (const float*)d_in[15];
  const float* rnn_w    = (const float*)d_in[16];
  const float* rnn_b    = (const float*)d_in[17];
  const float* attn1    = (const float*)d_in[18];
  const float* attn2    = (const float*)d_in[19];
  float* out = (float*)d_out;

  char* base = (char*)d_ws;
  size_t off = 0;
  auto alloc = [&](size_t bytes) -> void* {
    void* p = base + off;
    off = (off + bytes + 255) & ~(size_t)255;
    return p;
  };
  __bf16* featB   = (__bf16*)alloc((size_t)N_ * D_ * 2);
  __bf16* wqkvB   = (__bf16*)alloc((size_t)3 * D_ * D_ * 2);
  __bf16* woB     = (__bf16*)alloc((size_t)D_ * D_ * 2);
  __bf16* w1B     = (__bf16*)alloc((size_t)FF_ * D_ * 2);
  __bf16* w2B     = (__bf16*)alloc((size_t)D_ * FF_ * 2);
  __bf16* wrB     = (__bf16*)alloc((size_t)H_ * D_ * D_ * 2);
  float*  hidden  = (float*)alloc((size_t)E_ * D_ * 4);
  __bf16* hiddenB = (__bf16*)alloc((size_t)E_ * D_ * 2);
  float*  eft     = (float*)alloc((size_t)E_ * H_ * D_ * 4);
  float*  aE      = (float*)alloc((size_t)E_ * H_ * 4);
  float*  eExp    = (float*)alloc((size_t)E_ * H_ * 4);
  unsigned* mEnc  = (unsigned*)alloc((size_t)N_ * H_ * 4);
  float*  sN      = (float*)alloc((size_t)N_ * H_ * 4);
  float*  ft      = (float*)alloc((size_t)N_ * H_ * D_ * 4);

  // bf16 conversions (weights stay L2-resident afterwards)
  auto conv = [&](const float* s, __bf16* d, int n) {
    f2bf_kernel<<<(n + 255) / 256, 256, 0, stream>>>(s, d, n);
  };
  conv(features, featB, N_ * D_);
  conv(in_w,  wqkvB, 3 * D_ * D_);
  conv(out_w, woB,   D_ * D_);
  conv(w1,    w1B,   FF_ * D_);
  conv(w2,    w2B,   D_ * FF_);
  conv(rnn_w, wrB,   H_ * D_ * D_);

  // zero-init segment buffers (encoded -inf == 0 for mEnc)
  hipMemsetAsync(mEnc, 0, (size_t)N_ * H_ * 4, stream);
  hipMemsetAsync(sN,   0, (size_t)N_ * H_ * 4, stream);
  hipMemsetAsync(ft,   0, (size_t)N_ * H_ * D_ * 4, stream);

  encoder_kernel<<<E_ / 16, 256, 0, stream>>>(
      featB, emi, wqkvB, in_b, woB, out_b, ln1g, ln1b, w1B, b1, w2B, b2,
      ln2g, ln2b, hidden, hiddenB);

  rnn_kernel<<<E_ / 16, 256, 0, stream>>>(hiddenB, wrB, rnn_b, eft);

  attn_score_kernel<<<E_ * H_ / 8, 256, 0, stream>>>(features, emi, dst, eft,
                                                     attn1, attn2, aE, mEnc);
  exp_sum_kernel<<<(E_ * H_ + 255) / 256, 256, 0, stream>>>(aE, dst, mEnc, eExp, sN);
  scatter_kernel<<<E_ * H_ / 8, 256, 0, stream>>>(eExp, sN, dst, eft, ft);
  gather_out_kernel<<<T_ * 1024 / 256, 256, 0, stream>>>(ft, tgt, out);
}
